// ProjectConvBlock_82291573392032
// MI455X (gfx1250) — matrix operand order repaired and checked by reading, not compiled, not run on hardware
//
#include <hip/hip_runtime.h>
#include <hip/hip_bf16.h>

typedef __attribute__((ext_vector_type(16))) _Float16 v16h;
typedef __attribute__((ext_vector_type(4)))  _Float16 v4h;
typedef __attribute__((ext_vector_type(8)))  float    v8f;

#define CDIM 64
#define DD   64
#define HHH  128
#define WWW  128
#define SPATIAL (DD*HHH*WWW)          /* 1,048,576 */
#define KCH  192
#define NEPS 1e-5f
#define SLOPE 0.01f

#define CAT_HALVES ((size_t)KCH * (size_t)SPATIAL)   /* 201,326,592 halves = 402 MB */
#define STATS_BYTES 4096
#define ASWZ_HALVES (4*6*32*16)                      /* 12288 pre-swizzled A halves */

/* stats float offsets (within stats region after cat) */
#define OFF_CATSUM   0
#define OFF_CATSS    192
#define OFF_CATMU    384
#define OFF_CATRSTD  576
#define OFF_OUTSUM   768
#define OFF_OUTSS    832
#define OFF_OUTMU    896
#define OFF_OUTRSTD  960
#define STATS_FLOATS 1024

/* ------------------------------------------------------------------ */
__global__ __launch_bounds__(1024)
void k_zero_stats(float* __restrict__ stats) {
    int t = threadIdx.x;
    if (t < STATS_FLOATS) stats[t] = 0.0f;
}

/* ------------------------------------------------------------------ */
/* Pre-swizzle w_pw (f32 [64 x 192]) into f16 WMMA A-fragment order:  */
/* aswz[((mtile*6 + kstep)*32 + lane)*16 + h], so each lane's v16h is */
/* 32 contiguous bytes. A layout used here (contiguous k, B matches): m = mtile*16 + lane%16,         */
/* k = kstep*32 + h + 16*(lane>>4).                                   */
__global__ __launch_bounds__(256)
void k_prep_a(const float* __restrict__ wpw, _Float16* __restrict__ aswz) {
    int L = blockIdx.x * 256 + threadIdx.x;        /* 12288 total */
    int h = L & 15;
    int t = L >> 4;
    int lane  = t & 31;  t >>= 5;
    int kstep = t % 6;
    int mtile = t / 6;
    int m = mtile*16 + (lane & 15);
    int k = kstep*32 + h + 16*(lane >> 4);
    aswz[L] = (_Float16)wpw[m*KCH + k];
}

/* ------------------------------------------------------------------ */
/* Fused 3x depthwise conv: read x once per tile via LDS (with halo), */
/* write f16 pre-norm cat, accumulate per-channel sum / sumsq.        */
__global__ __launch_bounds__(256)
void k_dw(const float* __restrict__ x,
          const float* __restrict__ wd, const float* __restrict__ bd,
          const float* __restrict__ wh, const float* __restrict__ bh,
          const float* __restrict__ wwp, const float* __restrict__ bw,
          _Float16* __restrict__ cat, float* __restrict__ stats)
{
    const int TD = 2, TH = 8;
    __shared__ float tile[(TD+2)*(TH+2)*(WWW+2)];   /* 4*10*130 = 5200 f32 */
    __shared__ float sred[8];

    const int tid = threadIdx.x;
    const int h0  = blockIdx.x * TH;
    const int d0  = blockIdx.y * TD;
    const int c   = blockIdx.z;

    float w_d9[9], w_h9[9], w_w9[9];
#pragma unroll
    for (int i = 0; i < 9; ++i) {
        w_d9[i] = wd[c*9 + i];
        w_h9[i] = wh[c*9 + i];
        w_w9[i] = wwp[c*9 + i];
    }
    const float bdv = bd[c], bhv = bh[c], bwv = bw[c];
    const size_t cbase = (size_t)c * SPATIAL;

    for (int idx = tid; idx < 4*10*130; idx += 256) {
        int lw = idx % 130;
        int t  = idx / 130;
        int lh = t % 10;
        int ld = t / 10;
        int d = d0 + ld - 1, h = h0 + lh - 1, w = lw - 1;
        float v = 0.0f;
        if ((unsigned)d < (unsigned)DD && (unsigned)h < (unsigned)HHH &&
            (unsigned)w < (unsigned)WWW)
            v = x[cbase + (size_t)d*(HHH*WWW) + (size_t)h*WWW + w];
        tile[idx] = v;
    }
    __syncthreads();

    float s0=0.f,q0=0.f,s1=0.f,q1=0.f,s2=0.f,q2=0.f;
#pragma unroll
    for (int r = 0; r < 8; ++r) {
        int oidx = tid + r*256;             /* 2048 outputs per block */
        int ow = oidx % WWW;
        int t  = oidx / WWW;
        int oh = t % TH;
        int od = t / TH;
        int ld = od + 1, lh = oh + 1, lw = ow + 1;
        float yd = bdv, yh = bhv, yw = bwv;
#pragma unroll
        for (int i = 0; i < 3; ++i) {
#pragma unroll
            for (int j = 0; j < 3; ++j) {
                yd = fmaf(tile[(ld*10 + (lh+i-1))*130 + (lw+j-1)], w_d9[i*3+j], yd);
                yh = fmaf(tile[((ld+i-1)*10 + lh)*130 + (lw+j-1)], w_h9[i*3+j], yh);
                yw = fmaf(tile[((ld+i-1)*10 + (lh+j-1))*130 + lw], w_w9[i*3+j], yw);
            }
        }
        size_t s = (size_t)(d0+od)*(HHH*WWW) + (size_t)(h0+oh)*WWW + ow;
        _Float16 hd = (_Float16)yd, hh2 = (_Float16)yh, hw2 = (_Float16)yw;
        cat[(size_t)c       *SPATIAL + s] = hd;
        cat[(size_t)(64+c)  *SPATIAL + s] = hh2;
        cat[(size_t)(128+c) *SPATIAL + s] = hw2;
        float fd = (float)hd, fh = (float)hh2, fw = (float)hw2;
        s0 += fd; q0 += fd*fd;
        s1 += fh; q1 += fh*fh;
        s2 += fw; q2 += fw*fw;
    }

    float vals[6] = { s0, q0, s1, q1, s2, q2 };
#pragma unroll
    for (int v6 = 0; v6 < 6; ++v6) {
        float v = vals[v6];
        for (int off = 16; off > 0; off >>= 1) v += __shfl_xor(v, off, 32);
        if ((tid & 31) == 0) sred[tid >> 5] = v;
        __syncthreads();
        if (tid == 0) {
            float tot = 0.f;
            for (int i = 0; i < 8; ++i) tot += sred[i];
            int conv = v6 >> 1;
            float* dst = stats + ((v6 & 1) ? OFF_CATSS : OFF_CATSUM);
            atomicAdd(&dst[conv*64 + c], tot);
        }
        __syncthreads();
    }
}

/* ------------------------------------------------------------------ */
__global__ void k_cat_finalize(float* __restrict__ stats) {
    int k = threadIdx.x;
    if (k < KCH) {
        const float invS = 1.0f / (float)SPATIAL;
        float mu  = stats[OFF_CATSUM + k] * invS;
        float var = stats[OFF_CATSS  + k] * invS - mu*mu;
        stats[OFF_CATMU   + k] = mu;
        stats[OFF_CATRSTD + k] = rsqrtf(var + NEPS);
    }
}

/* ------------------------------------------------------------------ */
/* WMMA GEMM: out[m, s] = sum_k leaky(norm(cat[k, s])) * w_pw[m, k].  */
/* Block = 256 thr (8 waves): 4 m-tiles x 2 n-subtiles of 16 columns. */
__global__ __launch_bounds__(256)
void k_gemm(const _Float16* __restrict__ cat,
            const _Float16* __restrict__ aswz,
            const float* __restrict__ bpw,
            const float* __restrict__ stats, float* __restrict__ out)
{
    __shared__ _Float16 ldsB[6144] __attribute__((aligned(32))); /* frag-major */
    __shared__ float ldsMu[KCH], ldsR[KCH];

    const int tid = threadIdx.x;
    const size_t s0 = (size_t)blockIdx.x * 32;

    if (tid < KCH) {
        ldsMu[tid] = stats[OFF_CATMU   + tid];
        ldsR[tid]  = stats[OFF_CATRSTD + tid];
    }
    __syncthreads();

    /* Cooperative fill: normalize + LeakyReLU fused into B-operand load.
       v4h (8 B) global loads; stored so each lane's v16h fragment is 32
       contiguous bytes: [nsub][kstep][fragLane=khi*16+n%16][half=k%16]. */
    for (int i = tid; i < 1536; i += 256) {
        int k  = i >> 3;                /* 0..191 */
        int n4 = (i & 7) * 4;           /* 0,4,..,28 */
        v4h hv = *(const v4h*)&cat[(size_t)k * SPATIAL + s0 + n4];
        /* warm L2 for the next N-tile (global_prefetch_b8) */
        __builtin_prefetch(&cat[(size_t)k * SPATIAL + s0 + 32 + n4], 0, 1);
        float mu = ldsMu[k], rs = ldsR[k];
        int kstep = k >> 5, kk = k & 31;
        int klo   = kk & 15, khi = kk >> 4;
#pragma unroll
        for (int j = 0; j < 4; ++j) {
            int n = n4 + j;
            float y = ((float)hv[j] - mu) * rs;
            y = (y >= 0.0f) ? y : SLOPE * y;
            int nsub = n >> 4, nn = n & 15;
            ldsB[((((nsub*6 + kstep) * 32) + (khi*16 + nn)) * 16) + klo] =
                (_Float16)y;
        }
    }
    __syncthreads();

    const int wv     = tid >> 5;
    const int lane   = tid & 31;
    const int mtile  = wv & 3;
    const int nsub   = wv >> 2;
    const int laneHi = lane >> 4;
    const int lmod   = lane & 15;

    v8f acc = {};
#pragma unroll
    for (int kstep = 0; kstep < 6; ++kstep) {
        /* A fragment: one aligned v16h load from pre-swizzled f16 buffer */
        v16h a = *(const v16h*)&aswz[(((mtile*6 + kstep) * 32) + lane) * 16];
        /* B fragment: 32 contiguous bytes per lane from LDS */
        v16h b = *(const v16h*)&ldsB[(((nsub*6 + kstep) * 32) + lane) * 16];
        acc = __builtin_amdgcn_wmma_f32_16x16x32_f16(
                  false, a, false, b, (short)0, acc, false, false);
    }

    /* Epilogue: C/D layout — VGPR r: M = r + 8*laneHi, N = lane%16 */
#pragma unroll
    for (int r = 0; r < 8; ++r) {
        int mm = mtile*16 + r + 8*laneHi;
        size_t n = s0 + (size_t)nsub*16 + lmod;
        out[(size_t)mm * SPATIAL + n] = acc[r] + bpw[mm];
    }
}

/* ------------------------------------------------------------------ */
__global__ __launch_bounds__(256)
void k_out_stats(const float* __restrict__ out, float* __restrict__ stats)
{
    __shared__ float sred[8];
    const int m   = blockIdx.y;
    const int tid = threadIdx.x;
    const int CHUNK = SPATIAL / 64;     /* 16384 */
    const size_t base = (size_t)m * SPATIAL + (size_t)blockIdx.x * CHUNK;

    float s = 0.f, q = 0.f;
    for (int i = tid; i < CHUNK/4; i += 256) {
        const float4 v = *(const float4*)&out[base + (size_t)i*4];
        s += v.x + v.y + v.z + v.w;
        q += v.x*v.x + v.y*v.y + v.z*v.z + v.w*v.w;
    }
    float vals[2] = { s, q };
#pragma unroll
    for (int v2 = 0; v2 < 2; ++v2) {
        float v = vals[v2];
        for (int off = 16; off > 0; off >>= 1) v += __shfl_xor(v, off, 32);
        if ((tid & 31) == 0) sred[tid >> 5] = v;
        __syncthreads();
        if (tid == 0) {
            float tot = 0.f;
            for (int i = 0; i < 8; ++i) tot += sred[i];
            atomicAdd(&stats[(v2 ? OFF_OUTSS : OFF_OUTSUM) + m], tot);
        }
        __syncthreads();
    }
}

__global__ void k_out_finalize(float* __restrict__ stats) {
    int m = threadIdx.x;
    if (m < CDIM) {
        const float invS = 1.0f / (float)SPATIAL;
        float mu  = stats[OFF_OUTSUM + m] * invS;
        float var = stats[OFF_OUTSS  + m] * invS - mu*mu;
        stats[OFF_OUTMU   + m] = mu;
        stats[OFF_OUTRSTD + m] = rsqrtf(var + NEPS);
    }
}

__global__ __launch_bounds__(256)
void k_apply(float* __restrict__ out, const float* __restrict__ stats) {
    size_t e = ((size_t)blockIdx.x * 256 + threadIdx.x) * 4;
    int m = (int)(e >> 20);             /* SPATIAL = 2^20; float4 stays in-channel */
    float mu = stats[OFF_OUTMU + m], rs = stats[OFF_OUTRSTD + m];
    float4 v = *(const float4*)&out[e];
    float4 y;
    y.x = (v.x - mu) * rs; y.x = (y.x >= 0.f) ? y.x : SLOPE * y.x;
    y.y = (v.y - mu) * rs; y.y = (y.y >= 0.f) ? y.y : SLOPE * y.y;
    y.z = (v.z - mu) * rs; y.z = (y.z >= 0.f) ? y.z : SLOPE * y.z;
    y.w = (v.w - mu) * rs; y.w = (y.w >= 0.f) ? y.w : SLOPE * y.w;
    *(float4*)&out[e] = y;
}

/* ------------------------------------------------------------------ */
extern "C" void kernel_launch(void* const* d_in, const int* in_sizes, int n_in,
                              void* d_out, int out_size, void* d_ws, size_t ws_size,
                              hipStream_t stream)
{
    (void)in_sizes; (void)n_in; (void)out_size; (void)ws_size;

    const float* x   = (const float*)d_in[0];
    const float* wd  = (const float*)d_in[1];
    const float* bd  = (const float*)d_in[2];
    const float* wh  = (const float*)d_in[3];
    const float* bh  = (const float*)d_in[4];
    const float* wwp = (const float*)d_in[5];
    const float* bw  = (const float*)d_in[6];
    const float* wpw = (const float*)d_in[7];
    const float* bpw = (const float*)d_in[8];

    float*    out   = (float*)d_out;
    _Float16* cat   = (_Float16*)d_ws;
    float*    stats = (float*)((char*)d_ws + CAT_HALVES * sizeof(_Float16));
    _Float16* aswz  = (_Float16*)((char*)stats + STATS_BYTES);

    k_zero_stats <<<1, 1024, 0, stream>>>(stats);
    k_prep_a     <<<ASWZ_HALVES/256, 256, 0, stream>>>(wpw, aswz);
    k_dw         <<<dim3(HHH/8, DD/2, CDIM), 256, 0, stream>>>(
                     x, wd, bd, wh, bh, wwp, bw, cat, stats);
    k_cat_finalize<<<1, KCH, 0, stream>>>(stats);
    k_gemm       <<<SPATIAL/32, 256, 0, stream>>>(cat, aswz, bpw, stats, out);
    k_out_stats  <<<dim3(64, CDIM), 256, 0, stream>>>(out, stats);
    k_out_finalize<<<1, 64, 0, stream>>>(stats);
    k_apply      <<<(CDIM*(size_t)SPATIAL)/(256*4), 256, 0, stream>>>(out, stats);
}
